// GILRLSTMLayer_67980742362022
// MI455X (gfx1250) — matrix operand order repaired and checked by reading, not compiled, not run on hardware
//
#include <hip/hip_runtime.h>

// ---------------------------------------------------------------------------
// GILR-LSTM layer for MI455X (gfx1250)
//   - bf16 WMMA GEMMs (v_wmma_f32_16x16x32_bf16), f32 accumulate
//   - A tiles moved by CDNA5 async DMA: global_load_async_to_lds_b128
//     (ASYNCcnt-tracked, double-buffered LDS)
//   - weights pre-packed into B-fragment lane layout (coalesced b128 loads)
//   - fp32 linear-recurrence scans with fused gate math
// ---------------------------------------------------------------------------

typedef __attribute__((ext_vector_type(16))) __bf16 v16bf;
typedef __attribute__((ext_vector_type(8)))  float  v8f;

#define D_DIM    512       // hidden size (K and N of every GEMM)
#define M_DIM    32768     // B*T
#define KT_N     16        // 512 / 32 k-tiles
#define A_STRIDE 20        // LDS row stride in uint words (16 pairs + pad)

union FragBF {
  v16bf v;
  uint4 q[2];
};

__device__ __forceinline__ unsigned pack_bf16x2(float a, float b) {
  // round-to-nearest-even f32 -> bf16, pack {lo=a, hi=b}
  unsigned ua = __float_as_uint(a);
  unsigned ub = __float_as_uint(b);
  ua = (ua + 0x7FFFu + ((ua >> 16) & 1u)) >> 16;
  ub = (ub + 0x7FFFu + ((ub >> 16) & 1u)) & 0xFFFF0000u;
  return (ua & 0xFFFFu) | ub;
}

__device__ __forceinline__ unsigned short bf16_rne(float a) {
  unsigned ua = __float_as_uint(a);
  return (unsigned short)((ua + 0x7FFFu + ((ua >> 16) & 1u)) >> 16);
}

__device__ __forceinline__ float sigmoidf_(float x) {
  return 1.0f / (1.0f + __expf(-x));
}

// ---------------------------------------------------------------------------
// Kernel 0a: stream-convert x (f32) -> packed bf16 pairs (one uint = {2d,2d+1})
// ---------------------------------------------------------------------------
__global__ __launch_bounds__(256) void convert_x_kernel(
    const float* __restrict__ X, unsigned* __restrict__ out) {
  const size_t id = (size_t)blockIdx.x * 256 + threadIdx.x;  // over M*K/4
  const float4 v = ((const float4*)X)[id];
  uint2 r;
  r.x = pack_bf16x2(v.x, v.y);
  r.y = pack_bf16x2(v.z, v.w);
  ((uint2*)out)[id] = r;
}

// ---------------------------------------------------------------------------
// Kernel 0b: pack a (E, 512, 512) f32 weight tensor into WMMA B-fragment order.
// B (32x16 bf16, wave32): lane = 16*half + n, VGPR j = {W[k][n],W[k+1][n]},
// k = 16*(j/4) + 8*half + 2*(j%4).  Packed index: ((kt*32 + ntG)*32 + lane)*8 + j (uints).
// ---------------------------------------------------------------------------
__global__ __launch_bounds__(256) void pack_weights_kernel(
    const float* __restrict__ W, unsigned* __restrict__ out) {
  const int e  = blockIdx.y;
  const int id = blockIdx.x * 256 + threadIdx.x;   // 0 .. 131071
  const int j    = id & 7;
  const int lane = (id >> 3) & 31;
  const int nt   = (id >> 8) & 31;
  const int kt   = id >> 13;
  const int n = nt * 16 + (lane & 15);
  const int k = kt * 32 + ((j >> 2) << 4) + ((lane >> 4) << 3) + 2 * (j & 3);
  const float* Wm = W + (size_t)e * (D_DIM * D_DIM);
  out[(size_t)e * (KT_N * 32 * 32 * 8) + id] =
      pack_bf16x2(Wm[(size_t)k * D_DIM + n], Wm[(size_t)(k + 1) * D_DIM + n]);
}

// ---------------------------------------------------------------------------
// Tiled ensemble GEMM: C[e] = A(32768x512 bf16 pair-packed) * W[e](bf16 packed)
// Block: 256 threads = 8 waves; tile 128(M) x NT*16(N); K-step 32.
// A tiles DMA'd into double-buffered LDS with global_load_async_to_lds_b128;
// B fragments loaded straight from pre-packed weights (L2-resident).
// MODE 1: E=2 -> F = sigmoid(u1), G = (1-F)*tanh(u0)
// MODE 2: E=4 -> F = sigmoid(u0), G = (1-F)*sigmoid(u1)*tanh(u3), O = sigmoid(u2)
// MODE 3: E=1 -> out = u0 + bias
// ---------------------------------------------------------------------------
template<int E, int NT, int MODE>
__global__ __launch_bounds__(256) void gemm_ens_kernel(
    const unsigned* __restrict__ Abf,   // M x 256 packed bf16 pairs
    const unsigned* __restrict__ Wp,
    const float* __restrict__ bias,
    float* __restrict__ o0, float* __restrict__ o1, float* __restrict__ o2) {
  __shared__ __align__(16) unsigned Al[2][128 * A_STRIDE];

  const int tid   = threadIdx.x;
  const int wave  = tid >> 5;
  const int lane  = tid & 31;
  const int lhalf = lane >> 4;
  const int lcol  = lane & 15;
  const int m0 = blockIdx.y * 128;
  const int nb = blockIdx.x;                 // block of NT*16 columns
  const int n0 = nb * (NT * 16);

  const unsigned ldsBase0 = (unsigned)(uintptr_t)(&Al[0][0]);
  const unsigned ldsBase1 = ldsBase0 + 128 * A_STRIDE * 4;

  // Per-thread async transfer coordinates: 2 x b128 per thread per tile.
  // transfer id = tid + s*256 -> row = id>>2 (0..127), quarter q = id&3.
  auto issue_tile = [&](int kt2, unsigned ldsBase) {
#pragma unroll
    for (int s = 0; s < 2; ++s) {
      const int id  = tid + s * 256;
      const int row = id >> 2;
      const int q   = id & 3;
      const unsigned* src =
          Abf + (size_t)(m0 + row) * (D_DIM / 2) + kt2 * 16 + q * 4;
      const unsigned dst = ldsBase + (unsigned)(row * (A_STRIDE * 4) + q * 16);
      asm volatile("global_load_async_to_lds_b128 %0, %1, off"
                   :: "v"(dst), "v"(src) : "memory");
    }
  };

  v8f acc[E][NT];
  const v8f vzero = {0.f, 0.f, 0.f, 0.f, 0.f, 0.f, 0.f, 0.f};
#pragma unroll
  for (int e = 0; e < E; ++e)
#pragma unroll
    for (int nt = 0; nt < NT; ++nt) acc[e][nt] = vzero;

  issue_tile(0, ldsBase0);   // software pipeline: prefetch tiles 0 and 1
  issue_tile(1, ldsBase1);

  for (int kt = 0; kt < KT_N; ++kt) {
    // Async copies complete in order per wave: allowing 2 outstanding means
    // tile kt has landed (tile kt+1 may still be in flight).
    if (kt < KT_N - 1)
      asm volatile("s_wait_asynccnt 0x2" ::: "memory");
    else
      asm volatile("s_wait_asynccnt 0x0" ::: "memory");
    __syncthreads();

    // A fragment (16x32): two conflict-free b128 LDS reads (stride-20 pad).
    const unsigned* Acur = Al[kt & 1];
    FragBF af;
    const unsigned* ap = Acur + (wave * 16 + lcol) * A_STRIDE;
    af.q[0] = *(const uint4*)(ap + 4 * lhalf);
    af.q[1] = *(const uint4*)(ap + 8 + 4 * lhalf);

#pragma unroll
    for (int e = 0; e < E; ++e) {
      const unsigned* wp = Wp + (size_t)e * (KT_N * 32 * 32 * 8) +
                           ((((size_t)kt * 32) + (size_t)nb * NT) * 32 + lane) * 8;
#pragma unroll
      for (int nt = 0; nt < NT; ++nt) {
        FragBF bfrag;
        const unsigned* p = wp + (size_t)nt * 256;
        bfrag.q[0] = *(const uint4*)(p);
        bfrag.q[1] = *(const uint4*)(p + 4);
        acc[e][nt] = __builtin_amdgcn_wmma_f32_16x16x32_bf16(
            false, af.v, false, bfrag.v, (short)0, acc[e][nt], false, false);
      }
    }

    __syncthreads();                       // all waves done reading buf[kt&1]
    if (kt + 2 < KT_N)
      issue_tile(kt + 2, (kt & 1) ? ldsBase1 : ldsBase0);
  }

  // Epilogue: C/D layout -> VGPR r: lanes 0-15 M=r, lanes 16-31 M=r+8.
#pragma unroll
  for (int nt = 0; nt < NT; ++nt) {
    const int n = n0 + nt * 16 + lcol;
    float bv[E];
#pragma unroll
    for (int e = 0; e < E; ++e) bv[e] = bias[e * D_DIM + n];
#pragma unroll
    for (int r = 0; r < 8; ++r) {
      const int m = m0 + wave * 16 + 8 * lhalf + r;
      const size_t idx = (size_t)m * D_DIM + n;
      if constexpr (MODE == 1) {
        const float v = tanhf(acc[0][nt][r] + bv[0]);
        const float f = sigmoidf_(acc[1][nt][r] + bv[1]);
        o0[idx] = f;
        o1[idx] = (1.0f - f) * v;
      } else if constexpr (MODE == 2) {
        const float f2 = sigmoidf_(acc[0][nt][r] + bv[0]);
        const float ii = sigmoidf_(acc[1][nt][r] + bv[1]);
        const float oo = sigmoidf_(acc[2][nt][r] + bv[2]);
        const float zz = tanhf(acc[3][nt][r] + bv[3]);
        o0[idx] = f2;
        o1[idx] = (1.0f - f2) * ii * zz;
        o2[idx] = oo;
      } else {
        o0[idx] = acc[0][nt][r] + bv[0];
      }
    }
  }
}

// ---------------------------------------------------------------------------
// GILR scan: h_t = f_t * h_{t-1} + g_t over T=4096, one thread per (b,d).
// Recurrence kept in fp32; emits bf16 activations for the next WMMA GEMM.
// HASO: also multiplies by o on output (P = h*o). Last state -> hidden (f32).
// ---------------------------------------------------------------------------
template<bool HASO>
__global__ __launch_bounds__(256) void gilr_scan_kernel(
    const float* __restrict__ F, const float* __restrict__ G,
    const float* __restrict__ O, unsigned short* __restrict__ Hbf,
    float* __restrict__ hid, int hidOff) {
  const int gid = blockIdx.x * 256 + threadIdx.x;  // 0 .. 4095
  const int b = gid >> 9;
  const int d = gid & (D_DIM - 1);
  const size_t base = (size_t)b * 4096 * D_DIM + d;
  float h = 0.0f;
  for (int t0 = 0; t0 < 4096; t0 += 8) {
    float fv[8], gv[8], ov[8];
#pragma unroll
    for (int u = 0; u < 8; ++u) {
      const size_t idx = base + (size_t)(t0 + u) * D_DIM;
      fv[u] = F[idx];
      gv[u] = G[idx];
      if constexpr (HASO) ov[u] = O[idx];
    }
#pragma unroll
    for (int u = 0; u < 8; ++u) {
      h = fmaf(fv[u], h, gv[u]);
      const size_t idx = base + (size_t)(t0 + u) * D_DIM;
      if constexpr (HASO) Hbf[idx] = bf16_rne(h * ov[u]);
      else                Hbf[idx] = bf16_rne(h);
    }
  }
  hid[(size_t)b * 1024 + hidOff + d] = h;   // last (un-multiplied) state, f32
}

// ---------------------------------------------------------------------------
extern "C" void kernel_launch(void* const* d_in, const int* in_sizes, int n_in,
                              void* d_out, int out_size, void* d_ws, size_t ws_size,
                              hipStream_t stream) {
  (void)in_sizes; (void)n_in; (void)out_size; (void)ws_size;

  const float* x     = (const float*)d_in[0];   // (8,4096,512)
  const float* W_in  = (const float*)d_in[1];   // (2,512,512)
  const float* b_in  = (const float*)d_in[2];   // (2,512)
  const float* W_mid = (const float*)d_in[3];   // (4,512,512)
  const float* b_mid = (const float*)d_in[4];   // (4,512)
  const float* W_out = (const float*)d_in[5];   // (512,512)
  const float* b_out = (const float*)d_in[6];   // (512,)
  float* out = (float*)d_out;                   // out (8,4096,512) ++ hidden (1,8,1024)
  float* hid = out + (size_t)M_DIM * D_DIM;

  char* ws = (char*)d_ws;
  const size_t PACK_MAT = (size_t)KT_N * 32 * 32 * 8;       // uints per matrix
  unsigned* packIn  = (unsigned*)ws;                        // 2 matrices
  unsigned* packMid = packIn + 2 * PACK_MAT;                // 4 matrices
  unsigned* packOut = packMid + 4 * PACK_MAT;               // 1 matrix
  const size_t BUF = (size_t)M_DIM * D_DIM * sizeof(float); // 64 MB
  float* bufF = (float*)(ws + (4ull << 20));
  float* bufG = (float*)(ws + (4ull << 20) + BUF);
  float* bufO = (float*)(ws + (4ull << 20) + 2 * BUF);
  unsigned* xbf = (unsigned*)(ws + (4ull << 20) + 3 * BUF);           // 32 MB
  unsigned* abf = (unsigned*)(ws + (4ull << 20) + 3 * BUF + BUF / 2); // 32 MB (V, later P)

  // 0) convert x to packed bf16; pre-pack weights into B-fragment layout
  convert_x_kernel<<<dim3(M_DIM * D_DIM / 4 / 256), 256, 0, stream>>>(x, xbf);
  pack_weights_kernel<<<dim3(512, 2), 256, 0, stream>>>(W_in,  packIn);
  pack_weights_kernel<<<dim3(512, 4), 256, 0, stream>>>(W_mid, packMid);
  pack_weights_kernel<<<dim3(512, 1), 256, 0, stream>>>(W_out, packOut);

  // 1) u = x @ W_in + b_in ; F = sigmoid(u1), G = (1-F)*tanh(u0)
  gemm_ens_kernel<2, 4, 1><<<dim3(8, 256), 256, 0, stream>>>(
      xbf, packIn, b_in, bufF, bufG, nullptr);

  // 2) V = scan(F, G)  (bf16); hidden_pre -> hid[b][0:512]
  gilr_scan_kernel<false><<<16, 256, 0, stream>>>(
      bufF, bufG, nullptr, (unsigned short*)abf, hid, 0);

  // 3) u = V @ W_mid + b_mid ; F2/G2/O
  gemm_ens_kernel<4, 2, 2><<<dim3(16, 256), 256, 0, stream>>>(
      abf, packMid, b_mid, bufF, bufG, bufO);

  // 4) P = scan(F2, G2) * O  (bf16); hidden_middle -> hid[b][512:1024]
  gilr_scan_kernel<true><<<16, 256, 0, stream>>>(
      bufF, bufG, bufO, (unsigned short*)abf, hid, 512);

  // 5) out = P @ W_out + b_out
  gemm_ens_kernel<1, 4, 3><<<dim3(8, 256), 256, 0, stream>>>(
      abf, packOut, b_out, out, nullptr, nullptr);
}
